// head_attention_12034498363716
// MI455X (gfx1250) — matrix operand order repaired and checked by reading, not compiled, not run on hardware
//
#include <hip/hip_runtime.h>
#include <hip/hip_bf16.h>

typedef _Float16 half16 __attribute__((ext_vector_type(16)));
typedef float    float8 __attribute__((ext_vector_type(8)));
typedef float    float4v __attribute__((ext_vector_type(4)));

#define BB   4
#define NN   512
#define DD   128
#define DHH  64
#define KPAD 144   // padded LDS stride (halves) for 128-deep weight columns (288B: 32B-aligned, bank-staggered)
#define VPAD 528   // padded LDS stride (halves) for 512-deep V columns   (1056B: 32B-aligned, bank-staggered)

// Build the f16 A-operand for v_wmma_f32_16x16x32_f16 from an f32 row.
// Contiguous A layout (matches B): lane = 16*khalf + m; halves 0..15 hold
// K = khalf*16 + 0..15.
__device__ __forceinline__ half16 make_a16(const float* __restrict__ row, int kbase) {
  half16 a;
#pragma unroll
  for (int t = 0; t < 8; ++t) {
    a[t]     = (_Float16)row[kbase + t];
    a[t + 8] = (_Float16)row[kbase + 8 + t];
  }
  return a;
}

// Non-temporal variant (streaming data: e). Explicit float4 loads keep b128
// merging; kbase is always a multiple of 16 floats -> 64B aligned.
__device__ __forceinline__ half16 make_a16_nt(const float* __restrict__ row, int kbase) {
  const float4v* p0 = (const float4v*)(row + kbase);
  const float4v* p1 = (const float4v*)(row + kbase + 8);
  float4v c0 = __builtin_nontemporal_load(p0);
  float4v c1 = __builtin_nontemporal_load(p0 + 1);
  float4v c2 = __builtin_nontemporal_load(p1);
  float4v c3 = __builtin_nontemporal_load(p1 + 1);
  half16 a;
#pragma unroll
  for (int t = 0; t < 4; ++t) {
    a[t]      = (_Float16)c0[t];
    a[t + 4]  = (_Float16)c1[t];
    a[t + 8]  = (_Float16)c2[t];
    a[t + 12] = (_Float16)c3[t];
  }
  return a;
}

// B-operand: lane = 16*khalf + n holds 16 consecutive K values of column n.
__device__ __forceinline__ half16 load_b16(const _Float16* p) {
  return *(const half16*)p;
}

// ---------------------------------------------------------------------------
// Kernel 1: Q/K/V/Ni/Nj projections  (2048x128 @ 128x64, x5) via WMMA.
// grid = 16 blocks x 256 threads; each wave owns a 16-row M tile.
// ---------------------------------------------------------------------------
__global__ void __launch_bounds__(256) proj_kernel(
    const float* __restrict__ x, const float* __restrict__ nm,
    const float* __restrict__ Wq, const float* __restrict__ bq,
    const float* __restrict__ Wk, const float* __restrict__ bk,
    const float* __restrict__ Wv, const float* __restrict__ bv,
    const float* __restrict__ Wni, const float* __restrict__ bni,
    const float* __restrict__ Wnj, const float* __restrict__ bnj,
    float* __restrict__ Qo, float* __restrict__ Ko, float* __restrict__ Vo,
    float* __restrict__ Nio, float* __restrict__ Njo)
{
  __shared__ alignas(32) _Float16 sW[DHH * KPAD];   // one weight at a time, 18 KB

  const float* Ws[5]   = {Wq, Wk, Wv, Wni, Wnj};
  const float* bs[5]   = {bq, bk, bv, bni, bnj};
  float*       outs[5] = {Qo, Ko, Vo, Nio, Njo};

  const int lane = threadIdx.x & 31;
  const int wave = threadIdx.x >> 5;
  const int lo = lane & 15, hi = lane >> 4;
  const int row0 = blockIdx.x * 128 + wave * 16;
  const float8 fzero = {};

  // A tile (x rows) is invariant across the 5 weights: load once.
  const float* xrow = x + (size_t)(row0 + lo) * DD;
  half16 a[4];
#pragma unroll
  for (int kt = 0; kt < 4; ++kt) a[kt] = make_a16(xrow, kt * 32 + hi * 16);

  float mrow[8];
#pragma unroll
  for (int r = 0; r < 8; ++r) mrow[r] = nm[row0 + r + 8 * hi];

#pragma unroll
  for (int w = 0; w < 5; ++w) {
    __syncthreads();
    for (int idx = threadIdx.x; idx < DD * DHH; idx += 256) {
      int k = idx >> 6, c = idx & 63;
      sW[c * KPAD + k] = (_Float16)Ws[w][idx];
    }
    __syncthreads();

    float biasv[4];
#pragma unroll
    for (int nt = 0; nt < 4; ++nt) biasv[nt] = bs[w][lo + 16 * nt];

    float8 acc[4];
#pragma unroll
    for (int nt = 0; nt < 4; ++nt) {       // kt = 0 with inline-0 C
      half16 bm = load_b16(&sW[(lo + 16 * nt) * KPAD + hi * 16]);
      acc[nt] = __builtin_amdgcn_wmma_f32_16x16x32_f16(
          false, a[0], false, bm, (short)0, fzero, false, false);
    }
#pragma unroll
    for (int kt = 1; kt < 4; ++kt)
#pragma unroll
      for (int nt = 0; nt < 4; ++nt) {
        half16 bm = load_b16(&sW[(lo + 16 * nt) * KPAD + kt * 32 + hi * 16]);
        acc[nt] = __builtin_amdgcn_wmma_f32_16x16x32_f16(
            false, a[kt], false, bm, (short)0, acc[nt], false, false);
      }
#pragma unroll
    for (int r = 0; r < 8; ++r) {
      float m = (w < 3) ? mrow[r] : 1.0f;   // Q,K,V masked; Ni,Nj raw (+bias)
      float* orow = outs[w] + (size_t)(row0 + r + 8 * hi) * DHH;
#pragma unroll
      for (int nt = 0; nt < 4; ++nt)
        orow[lo + 16 * nt] = (acc[nt][r] + biasv[nt]) * m;
    }
  }
}

// ---------------------------------------------------------------------------
// Kernel 2: fused  e_out = (e@We + be + Ni + Nj)*mask  and  Att logits.
// One block per (b,i); each wave sweeps 16-j M tiles (A tile is a contiguous
// 16x128 f32 chunk of e, loaded non-temporally). Att row reduced with
// shfl_xor inside 16-lane groups (wave32).
// ---------------------------------------------------------------------------
__global__ void __launch_bounds__(256) fused_e_kernel(
    const float* __restrict__ e, const float* __restrict__ nm,
    const float* __restrict__ We, const float* __restrict__ be,
    const float* __restrict__ Q, const float* __restrict__ K,
    const float* __restrict__ Ni, const float* __restrict__ Nj,
    float* __restrict__ att, float* __restrict__ eo)
{
  __shared__ alignas(32) _Float16 sW[DHH * KPAD];    // We, f16 column-major
  for (int idx = threadIdx.x; idx < DD * DHH; idx += 256) {
    int k = idx >> 6, c = idx & 63;
    sW[c * KPAD + k] = (_Float16)We[idx];
  }
  __syncthreads();

  const int bi = blockIdx.x;          // b*N + i
  const int b  = bi >> 9;
  const int lane = threadIdx.x & 31;
  const int wave = threadIdx.x >> 5;
  const int lo = lane & 15, hi = lane >> 4;
  const float8 fzero = {};

  const float mi = nm[bi];
  const float* qrow  = Q  + (size_t)bi * DHH;
  const float* nirow = Ni + (size_t)bi * DHH;
  float qv[4], bniv[4];               // bniv = be[c] + Ni[c]: per-column constant
#pragma unroll
  for (int nt = 0; nt < 4; ++nt) {
    int c = lo + 16 * nt;
    qv[nt] = qrow[c];
    bniv[nt] = be[c] + nirow[c];
  }

  for (int jt = wave; jt < 32; jt += 8) {
    const int j0 = jt * 16;
    const float* erow = e + ((size_t)bi * NN + (j0 + lo)) * DD;
    if (jt + 8 < 32)
      __builtin_prefetch(e + ((size_t)bi * NN + (j0 + 128 + lo)) * DD, 0, 1);

    half16 a[4];
#pragma unroll
    for (int kt = 0; kt < 4; ++kt) a[kt] = make_a16_nt(erow, kt * 32 + hi * 16);

    float8 acc[4];
#pragma unroll
    for (int nt = 0; nt < 4; ++nt) {       // kt = 0 with inline-0 C
      half16 bm = load_b16(&sW[(lo + 16 * nt) * KPAD + hi * 16]);
      acc[nt] = __builtin_amdgcn_wmma_f32_16x16x32_f16(
          false, a[0], false, bm, (short)0, fzero, false, false);
    }
#pragma unroll
    for (int kt = 1; kt < 4; ++kt)
#pragma unroll
      for (int nt = 0; nt < 4; ++nt) {
        half16 bm = load_b16(&sW[(lo + 16 * nt) * KPAD + kt * 32 + hi * 16]);
        acc[nt] = __builtin_amdgcn_wmma_f32_16x16x32_f16(
            false, a[kt], false, bm, (short)0, acc[nt], false, false);
      }

    // D-tile lane map: element (m = r + 8*hi, n = lo) of N-tile nt → col 16*nt+lo.
    float myatt = 0.0f;
#pragma unroll
    for (int r = 0; r < 8; ++r) {
      const int j  = j0 + r + 8 * hi;
      const int bj = (b << 9) + j;
      const float* krow  = K  + (size_t)bj * DHH;
      const float* njrow = Nj + (size_t)bj * DHH;
      const float mij = mi * nm[bj];
      float* erow_out = eo + ((size_t)bi * NN + j) * DHH;
      float s = 0.0f;
#pragma unroll
      for (int nt = 0; nt < 4; ++nt) {
        int c = lo + 16 * nt;
        float v = (acc[nt][r] + bniv[nt] + njrow[c]) * mij;   // e_out
        s += qv[nt] * krow[c] * v;                            // Q*e_out*K
        __builtin_nontemporal_store(v, &erow_out[c]);
      }
      // sum over the 64 columns: reduce across the 16-lane group (wave32)
      s += __shfl_xor(s, 1, 32);
      s += __shfl_xor(s, 2, 32);
      s += __shfl_xor(s, 4, 32);
      s += __shfl_xor(s, 8, 32);
      float av = (mij > 0.0f) ? s * 0.125f : -1.0e9f;         // /sqrt(64), mask
      if (lo == r) myatt = av;                                // lane lo keeps j=j0+lo+8*hi
    }
    if (lo < 8) att[(size_t)bi * NN + j0 + lo + 8 * hi] = myatt;
  }
}

// ---------------------------------------------------------------------------
// Kernel 3: softmax over axis=1 (i), in place. Lanes = consecutive j (coalesced).
// grid = B * (N/32) = 64 blocks of 256 = 8(i-strip) x 32(j).
// ---------------------------------------------------------------------------
__global__ void __launch_bounds__(256) softmax_kernel(float* __restrict__ att) {
  const int b  = blockIdx.x >> 4;
  const int tj = threadIdx.x & 31;
  const int ti = threadIdx.x >> 5;
  const int j  = ((blockIdx.x & 15) << 5) + tj;
  float* col = att + (size_t)b * NN * NN + j;
  __shared__ float red[8][32];

  float mx = -1.0e30f;
  for (int i = ti; i < NN; i += 8) mx = fmaxf(mx, col[(size_t)i * NN]);
  red[ti][tj] = mx;
  __syncthreads();
  if (ti == 0) {
#pragma unroll
    for (int t = 1; t < 8; ++t) mx = fmaxf(mx, red[t][tj]);
    red[0][tj] = mx;
  }
  __syncthreads();
  mx = red[0][tj];
  __syncthreads();

  float sum = 0.0f;
  for (int i = ti; i < NN; i += 8) sum += __expf(col[(size_t)i * NN] - mx);
  red[ti][tj] = sum;
  __syncthreads();
  if (ti == 0) {
#pragma unroll
    for (int t = 1; t < 8; ++t) sum += red[t][tj];
    red[0][tj] = sum;
  }
  __syncthreads();
  const float inv = 1.0f / red[0][tj];
  for (int i = ti; i < NN; i += 8) {
    size_t off = (size_t)i * NN;
    col[off] = __expf(col[off] - mx) * inv;
  }
}

// ---------------------------------------------------------------------------
// Kernel 4: x_out = Att @ V (512x512x64 per batch) via WMMA, masked.
// grid = 16 blocks x 8 waves; each wave owns one 16-row i tile, K=512.
// ---------------------------------------------------------------------------
__global__ void __launch_bounds__(256) xout_kernel(
    const float* __restrict__ att, const float* __restrict__ V,
    const float* __restrict__ nm, float* __restrict__ xo)
{
  __shared__ alignas(32) _Float16 sV[DHH * VPAD];   // V^T in f16, ~66 KB
  const int b = blockIdx.x >> 2;
  const float* Vb = V + (size_t)b * NN * DHH;
  for (int idx = threadIdx.x; idx < NN * DHH; idx += 256) {
    int jj = idx >> 6, c = idx & 63;
    sV[c * VPAD + jj] = (_Float16)Vb[idx];
  }
  __syncthreads();

  const int lane = threadIdx.x & 31;
  const int wave = threadIdx.x >> 5;
  const int lo = lane & 15, hi = lane >> 4;
  const int i0 = ((blockIdx.x & 3) * 8 + wave) * 16;
  const float8 fzero = {};

  const float* prow = att + ((size_t)b * NN + i0 + lo) * NN;
  float8 acc[4];
  {
    half16 a = make_a16(prow, hi * 16);              // kt = 0 with inline-0 C
#pragma unroll
    for (int nt = 0; nt < 4; ++nt) {
      half16 bm = load_b16(&sV[(lo + 16 * nt) * VPAD + hi * 16]);
      acc[nt] = __builtin_amdgcn_wmma_f32_16x16x32_f16(
          false, a, false, bm, (short)0, fzero, false, false);
    }
  }
  for (int kt = 1; kt < 16; ++kt) {
    half16 a = make_a16(prow, kt * 32 + hi * 16);
#pragma unroll
    for (int nt = 0; nt < 4; ++nt) {
      half16 bm = load_b16(&sV[(lo + 16 * nt) * VPAD + kt * 32 + hi * 16]);
      acc[nt] = __builtin_amdgcn_wmma_f32_16x16x32_f16(
          false, a, false, bm, (short)0, acc[nt], false, false);
    }
  }
#pragma unroll
  for (int r = 0; r < 8; ++r) {
    const int i = i0 + r + 8 * hi;
    const float m = nm[(b << 9) + i];
    float* orow = xo + (size_t)((b << 9) + i) * DHH;
#pragma unroll
    for (int nt = 0; nt < 4; ++nt) orow[lo + 16 * nt] = acc[nt][r] * m;
  }
}

// ---------------------------------------------------------------------------
extern "C" void kernel_launch(void* const* d_in, const int* in_sizes, int n_in,
                              void* d_out, int out_size, void* d_ws, size_t ws_size,
                              hipStream_t stream) {
  (void)in_sizes; (void)n_in; (void)out_size; (void)ws_size;
  const float* x   = (const float*)d_in[0];
  const float* e   = (const float*)d_in[1];
  const float* nm  = (const float*)d_in[2];
  const float* Wq  = (const float*)d_in[3];  const float* bq  = (const float*)d_in[4];
  const float* Wk  = (const float*)d_in[5];  const float* bk  = (const float*)d_in[6];
  const float* We  = (const float*)d_in[7];  const float* be  = (const float*)d_in[8];
  const float* Wv  = (const float*)d_in[9];  const float* bv  = (const float*)d_in[10];
  const float* Wni = (const float*)d_in[11]; const float* bni = (const float*)d_in[12];
  const float* Wnj = (const float*)d_in[13]; const float* bnj = (const float*)d_in[14];

  float* xo = (float*)d_out;                       // x_out: B*N*DH
  float* eo = xo + (size_t)BB * NN * DHH;          // e_out: B*N*N*DH

  const size_t P = (size_t)BB * NN * DHH;          // 131072 floats
  float* ws  = (float*)d_ws;
  float* Q   = ws;
  float* K   = Q + P;
  float* V   = K + P;
  float* Ni  = V + P;
  float* Nj  = Ni + P;
  float* att = Nj + P;                             // B*N*N floats

  proj_kernel<<<16, 256, 0, stream>>>(x, nm, Wq, bq, Wk, bk, Wv, bv,
                                      Wni, bni, Wnj, bnj, Q, K, V, Ni, Nj);
  fused_e_kernel<<<BB * NN, 256, 0, stream>>>(e, nm, We, be, Q, K, Ni, Nj, att, eo);
  softmax_kernel<<<64, 256, 0, stream>>>(att);
  xout_kernel<<<16, 256, 0, stream>>>(att, V, nm, xo);
}
